// ClsHeadModelFERDA_16372415332666
// MI455X (gfx1250) — matrix operand order repaired and checked by reading, not compiled, not run on hardware
//
#include <hip/hip_runtime.h>
#include <hip/hip_bf16.h>
#include <stdint.h>
#include <stddef.h>

#define B_  2
#define S_  2048
#define D_  768
#define H_  12
#define HD_ 64
#define W1_ 256
#define NC_ 8
#define FF_ 3072
#define NL_ 2
#define M_  (B_*S_)   // 4096 tokens per encoder pass

typedef unsigned short bfu;
typedef __attribute__((ext_vector_type(4)))  unsigned int u32x4;
typedef __attribute__((ext_vector_type(8)))  __bf16       v8bf;
typedef __attribute__((ext_vector_type(16))) __bf16       v16bf;
typedef __attribute__((ext_vector_type(8)))  float        v8f;

// TDM descriptor vector types (per CDNA5_HIP.md probe: ROCm7.2 = 5 args)
typedef __attribute__((ext_vector_type(4))) unsigned int tdm_u4;
typedef __attribute__((ext_vector_type(8))) int          tdm_i8;
typedef __attribute__((ext_vector_type(4))) int          tdm_i4;

#if __has_builtin(__builtin_amdgcn_tensor_load_to_lds) && \
    __has_builtin(__builtin_amdgcn_s_wait_tensorcnt)
#define HAVE_TDM 1
__device__ __forceinline__ void tdm_load(tdm_u4 g0, tdm_i8 g1) {
  tdm_i4 z4 = {};
#if __clang_major__ >= 23
  tdm_i8 z8 = {};
  __builtin_amdgcn_tensor_load_to_lds(g0, g1, z4, z4, z8, 0);
#else
  __builtin_amdgcn_tensor_load_to_lds(g0, g1, z4, z4, 0);
#endif
}
#endif

__device__ __forceinline__ bfu f2bf(float f) {
  unsigned u = __builtin_bit_cast(unsigned, f);
  u += 0x7FFFu + ((u >> 16) & 1u);          // round-to-nearest-even
  return (bfu)(u >> 16);
}
__device__ __forceinline__ float bf2f(bfu h) {
  unsigned u = ((unsigned)h) << 16;
  return __builtin_bit_cast(float, u);
}
__device__ __forceinline__ v16bf cat16(u32x4 lo, u32x4 hi) {
  v8bf a = __builtin_bit_cast(v8bf, lo);
  v8bf b = __builtin_bit_cast(v8bf, hi);
  return __builtin_shufflevector(a, b, 0,1,2,3,4,5,6,7,8,9,10,11,12,13,14,15);
}

// ---------------------------------------------------------------------------
// fp32 -> bf16 converter (weight pre-pass)
// ---------------------------------------------------------------------------
__global__ void cvt_bf16_k(const float* __restrict__ in, bfu* __restrict__ out, int n) {
  int i = blockIdx.x * 256 + threadIdx.x;
  if (i < n) out[i] = f2bf(in[i]);
}

// ---------------------------------------------------------------------------
// bf16 WMMA GEMM: out = act((A[MxK] @ W[KxN] + bias) * oscale)
// 256 threads = 8 waves (4x2), workgroup tile 128x64, BK=32.
// A tile staged to LDS via the Tensor Data Mover (wave 0 issues the descriptor,
// drains TENSORcnt); B staged transposed [n][k] so B-frag lanes read two
// 8-element K runs (ISA 16-bit B layout).
// ---------------------------------------------------------------------------
struct GemmSmem {
  alignas(16) bfu As[128][32];   // offset 0   (TDM destination)
  alignas(16) bfu Bs[64][32];    // offset 8192
};

__global__ __launch_bounds__(256) void gemm_bf16_wmma(
    const bfu* __restrict__ A, const bfu* __restrict__ W,
    const float* __restrict__ bias,
    float* __restrict__ outF, bfu* __restrict__ outH,
    int M, int N, int K, float oscale, int act)
{
  __shared__ GemmSmem sm;
  const int tid  = threadIdx.x;
  const int lane = tid & 31, wave = tid >> 5;
  const int wm = wave >> 1, wn = wave & 1;      // 4 x 2 waves of 32x32 tiles
  const int m0 = blockIdx.y * 128, n0 = blockIdx.x * 64;

  v8f acc[2][2] = {};
  for (int k0 = 0; k0 < K; k0 += 32) {
#ifdef HAVE_TDM
    // --- stage A via TDM: 128-row x 32-col bf16 tile, row stride K elements
    if (wave == 0) {
      unsigned long long ga = (unsigned long long)(const void*)&A[(size_t)m0 * K + k0];
      unsigned ldsoff = (unsigned)(size_t)(void*)&sm.As[0][0];
      tdm_u4 g0;
      g0[0] = 1u;                                        // count=1 (valid user D#)
      g0[1] = ldsoff;                                    // lds_addr (bytes)
      g0[2] = (unsigned)(ga & 0xFFFFFFFFu);              // global_addr[31:0]
      g0[3] = (unsigned)((ga >> 32) & 0x1FFFFFFu)        // global_addr[56:32]
              | 0x80000000u;                             // type = 2 ("image")
      tdm_i8 g1;
      g1[0] = 1 << 16;                                   // data_size=1 (2B), wg_mask=0
      g1[1] = (int)(((unsigned)K & 0xFFFFu) << 16);      // tensor_dim0[15:0]
      g1[2] = (int)(((unsigned)K >> 16) |
                    (((unsigned)M & 0xFFFFu) << 16));    // tensor_dim0[31:16], dim1[15:0]
      g1[3] = (int)(((unsigned)M >> 16) | (32u << 16));  // dim1[31:16], tile_dim0=32
      g1[4] = 128;                                       // tile_dim1=128, tile_dim2=0
      g1[5] = K;                                         // tensor_dim0_stride[31:0]
      g1[6] = 0;                                         // stride[47:32], dim1_stride lo
      g1[7] = 0;
      tdm_load(g0, g1);
    }
#else
    // --- fallback: per-lane staging of A: 128x32 bf16 (two b128 loads/thread)
#pragma unroll
    for (int i = 0; i < 2; ++i) {
      int c = tid + 256 * i;
      int r = c >> 2, cc = (c & 3) * 8;
      *(u32x4*)&sm.As[r][cc] = *(const u32x4*)&A[(size_t)(m0 + r) * K + k0 + cc];
    }
#endif
    // --- stage B transposed: Bs[n][k]
    {
      int k = tid >> 3, nc = (tid & 7) * 8;
      u32x4 w8 = *(const u32x4*)&W[(size_t)(k0 + k) * N + n0 + nc];
      const bfu* ws = (const bfu*)&w8;
#pragma unroll
      for (int j = 0; j < 8; ++j) sm.Bs[nc + j][k] = ws[j];
    }
    if (k0 + 32 < K)
      __builtin_prefetch(&W[(size_t)(k0 + 32 + (tid >> 3)) * N + n0], 0, 1);
#ifdef HAVE_TDM
    if (wave == 0) __builtin_amdgcn_s_wait_tensorcnt(0);
#endif
    __syncthreads();

    const int klA = (lane & 16) ? 8 : 0;     // A: lanes 0-15 K0-7/16-23, 16-31 K8-15/24-31
    const int klB = (lane & 16) ? 8 : 0;     // B: lanes 0-15 K0-7/16-23, 16-31 K8-15/24-31
    v16bf af[2], bfr[2];
#pragma unroll
    for (int mi = 0; mi < 2; ++mi) {
      int m = wm * 32 + mi * 16 + (lane & 15);
      af[mi] = cat16(*(const u32x4*)&sm.As[m][klA], *(const u32x4*)&sm.As[m][klA + 16]);
    }
#pragma unroll
    for (int ni = 0; ni < 2; ++ni) {
      int n = wn * 32 + ni * 16 + (lane & 15);
      bfr[ni] = cat16(*(const u32x4*)&sm.Bs[n][klB], *(const u32x4*)&sm.Bs[n][klB + 16]);
    }
#pragma unroll
    for (int mi = 0; mi < 2; ++mi)
#pragma unroll
      for (int ni = 0; ni < 2; ++ni)
        acc[mi][ni] = __builtin_amdgcn_wmma_f32_16x16x32_bf16(
            false, af[mi], false, bfr[ni], (short)0, acc[mi][ni], false, false);
    __syncthreads();
  }
  // epilogue: bias + scale + activation, fp32 and/or bf16 outputs
#pragma unroll
  for (int mi = 0; mi < 2; ++mi) {
#pragma unroll
    for (int ni = 0; ni < 2; ++ni) {
      int col = n0 + wn * 32 + ni * 16 + (lane & 15);
      float bcol = bias ? bias[col] : 0.f;
#pragma unroll
      for (int r = 0; r < 8; ++r) {
        int row = m0 + wm * 32 + mi * 16 + r + ((lane & 16) ? 8 : 0);
        float v = (acc[mi][ni][r] + bcol) * oscale;
        if (act == 1) v = 0.5f * v * (1.0f + erff(v * 0.70710678118f));  // exact GELU
        if (outF) outF[(size_t)row * N + col] = v;
        if (outH) outH[(size_t)row * N + col] = f2bf(v);
      }
    }
  }
}

// ---------------------------------------------------------------------------
// Sliding-window attention: one workgroup per (b, chunk, 32-row q-tile, head).
// Window of 768 keys processed in two 384-key halves through one LDS buffer.
// ---------------------------------------------------------------------------
__global__ __launch_bounds__(256) void sliding_attn_k(
    const bfu* __restrict__ qB, const bfu* __restrict__ kB, const bfu* __restrict__ vB,
    const int* __restrict__ x, int enc, bfu* __restrict__ ctxB)
{
  extern __shared__ char smem[];
  float (*sc)[3 * W1_] = (float(*)[3 * W1_])smem;                        // 32 x 768 f32
  bfu   (*pr)[3 * W1_] = (bfu(*)[3 * W1_])(smem + 32 * 768 * 4);         // 32 x 768 bf16
  bfu*  kvraw          = (bfu*)(smem + 32 * 768 * 4 + 32 * 768 * 2);     // 384 x 64 bf16
  bfu   (*qs)[HD_]     = (bfu(*)[HD_])(smem + 32 * 768 * 4 + 32 * 768 * 2 + 384 * 64 * 2);

  int blk = blockIdx.x;
  int h  = blk % H_;  blk /= H_;
  int qt = blk % 8;   blk /= 8;
  int c  = blk % NC_; blk /= NC_;
  int b  = blk;

  const int tid = threadIdx.x, lane = tid & 31, wave = tid >> 5;
  const int srow0 = c * W1_ + qt * 32;                 // global seq row of q tile
  const int maskbase = (b * 6 + 3 * enc + 1) * S_;

  // q tile 32x64 into LDS
  {
    int r = tid >> 3, dc = (tid & 7) * 8;
    *(u32x4*)&qs[r][dc] = *(const u32x4*)&qB[((size_t)(b * S_ + srow0 + r)) * D_ + h * HD_ + dc];
  }

  // phase 1: scores = q @ k^T, 32 x 768, WMMA over HD=64 in two k-steps
  bfu (*kk)[HD_] = (bfu(*)[HD_])kvraw;                 // [key 384][hd 64]
  for (int hh = 0; hh < 2; ++hh) {
    __syncthreads();
#pragma unroll
    for (int i = 0; i < 12; ++i) {
      int cc = tid + 256 * i;
      int key = cc >> 3, dc = (cc & 7) * 8;
      int p = c * W1_ - W1_ + hh * 384 + key;
      u32x4 val = {};
      if (p >= 0 && p < S_)
        val = *(const u32x4*)&kB[((size_t)(b * S_ + p)) * D_ + h * HD_ + dc];
      *(u32x4*)&kk[key][dc] = val;
    }
    __syncthreads();
    const int klA = (lane & 16) ? 8 : 0;
    const int klB = (lane & 16) ? 8 : 0;
    for (int idx = 0; idx < 6; ++idx) {
      int t2 = wave * 6 + idx;                         // 48 16x16 tiles / 8 waves
      int mt = t2 & 1, nt = t2 >> 1;                   // nt in 0..23
      v8f accs = {};
#pragma unroll
      for (int ks = 0; ks < 2; ++ks) {
        int m = mt * 16 + (lane & 15);
        v16bf a = cat16(*(const u32x4*)&qs[m][ks * 32 + klA],
                        *(const u32x4*)&qs[m][ks * 32 + klA + 16]);
        int n = nt * 16 + (lane & 15);
        v16bf bb = cat16(*(const u32x4*)&kk[n][ks * 32 + klB],
                         *(const u32x4*)&kk[n][ks * 32 + klB + 16]);
        accs = __builtin_amdgcn_wmma_f32_16x16x32_bf16(false, a, false, bb,
                                                       (short)0, accs, false, false);
      }
#pragma unroll
      for (int r = 0; r < 8; ++r) {
        int row = mt * 16 + r + ((lane & 16) ? 8 : 0);
        sc[row][hh * 384 + nt * 16 + (lane & 15)] = accs[r];
      }
    }
  }
  __syncthreads();

  // phase 2: band + key masking, row softmax (wave handles 4 rows, 24 cols/lane)
  for (int rr = 0; rr < 4; ++rr) {
    int row = wave * 4 + rr;
    int ql = qt * 32 + row;                            // q position within chunk
    float vals[24];
    float mx = -1e30f;
#pragma unroll
    for (int t = 0; t < 24; ++t) {
      int j = lane + 32 * t;
      int p = c * W1_ - W1_ + j;
      int dlt = ql - (j - W1_);
      bool ok = (dlt <= W1_) && (dlt >= -W1_) && (p >= 0) && (p < S_) &&
                (x[maskbase + p] > 0);
      float sv = ok ? sc[row][j] : -1e9f;
      vals[t] = sv;
      mx = fmaxf(mx, sv);
    }
    for (int off = 16; off; off >>= 1) mx = fmaxf(mx, __shfl_xor(mx, off, 32));
    float sum = 0.f;
#pragma unroll
    for (int t = 0; t < 24; ++t) { vals[t] = __expf(vals[t] - mx); sum += vals[t]; }
    for (int off = 16; off; off >>= 1) sum += __shfl_xor(sum, off, 32);
    float inv = 1.0f / sum;
#pragma unroll
    for (int t = 0; t < 24; ++t) pr[row][lane + 32 * t] = f2bf(vals[t] * inv);
  }

  // phase 3: ctx = probs @ vwin (32x64), V staged transposed [hd][key]
  bfu (*vt)[384] = (bfu(*)[384])kvraw;
  int mt = wave & 1, nt = wave >> 1;                   // 2 m-tiles x 4 n-tiles, 1/wave
  v8f accc = {};
  for (int hh = 0; hh < 2; ++hh) {
    __syncthreads();
#pragma unroll
    for (int i = 0; i < 12; ++i) {
      int cc = tid + 256 * i;
      int key = cc >> 3, dc = (cc & 7) * 8;
      int p = c * W1_ - W1_ + hh * 384 + key;
      u32x4 val = {};
      if (p >= 0 && p < S_)
        val = *(const u32x4*)&vB[((size_t)(b * S_ + p)) * D_ + h * HD_ + dc];
      const bfu* vv = (const bfu*)&val;
#pragma unroll
      for (int j = 0; j < 8; ++j) vt[dc + j][key] = vv[j];
    }
    __syncthreads();
    const int klA = (lane & 16) ? 8 : 0;
    const int klB = (lane & 16) ? 8 : 0;
    for (int ks = 0; ks < 12; ++ks) {
      int m = mt * 16 + (lane & 15);
      int kb = hh * 384 + ks * 32;
      v16bf a = cat16(*(const u32x4*)&pr[m][kb + klA],
                      *(const u32x4*)&pr[m][kb + klA + 16]);
      int n = nt * 16 + (lane & 15);
      v16bf bb = cat16(*(const u32x4*)&vt[n][ks * 32 + klB],
                       *(const u32x4*)&vt[n][ks * 32 + klB + 16]);
      accc = __builtin_amdgcn_wmma_f32_16x16x32_bf16(false, a, false, bb,
                                                     (short)0, accc, false, false);
    }
  }
#pragma unroll
  for (int r = 0; r < 8; ++r) {
    int row = mt * 16 + r + ((lane & 16) ? 8 : 0);
    int s = srow0 + row;
    int hd = nt * 16 + (lane & 15);
    ctxB[((size_t)(b * S_ + s)) * D_ + h * HD_ + hd] = f2bf(accc[r]);
  }
}

// ---------------------------------------------------------------------------
// Embedding gather + LayerNorm (one block per token, 3 elems/thread)
// ---------------------------------------------------------------------------
__global__ __launch_bounds__(256) void embed_ln_k(
    const int* __restrict__ x, int enc,
    const float* __restrict__ we, const float* __restrict__ pe, const float* __restrict__ te,
    const float* __restrict__ g, const float* __restrict__ bta,
    float* __restrict__ hF, bfu* __restrict__ hB)
{
  __shared__ float red[256], red2[256];
  __shared__ float s_mean, s_rstd;
  int token = blockIdx.x;
  int b = token >> 11, s = token & 2047;
  int id = x[(b * 6 + 3 * enc + 0) * S_ + s];
  int ty = x[(b * 6 + 3 * enc + 2) * S_ + s];
  int tid = threadIdx.x;
  float e[3]; float sum = 0.f, sq = 0.f;
#pragma unroll
  for (int i = 0; i < 3; ++i) {
    int d = tid + 256 * i;
    float v = we[(size_t)id * D_ + d] + pe[(size_t)(s + 2) * D_ + d] + te[(size_t)ty * D_ + d];
    e[i] = v; sum += v; sq += v * v;
  }
  red[tid] = sum; red2[tid] = sq; __syncthreads();
  for (int st = 128; st; st >>= 1) {
    if (tid < st) { red[tid] += red[tid + st]; red2[tid] += red2[tid + st]; }
    __syncthreads();
  }
  if (tid == 0) {
    float m = red[0] / D_;
    float var = red2[0] / D_ - m * m;
    s_mean = m; s_rstd = rsqrtf(var + 1e-5f);
  }
  __syncthreads();
  float m = s_mean, rs = s_rstd;
#pragma unroll
  for (int i = 0; i < 3; ++i) {
    int d = tid + 256 * i;
    float v = (e[i] - m) * rs * g[d] + bta[d];
    hF[(size_t)token * D_ + d] = v;
    hB[(size_t)token * D_ + d] = f2bf(v);
  }
}

// ---------------------------------------------------------------------------
// Residual add + LayerNorm (in-place safe: reads before barrier, writes after)
// ---------------------------------------------------------------------------
__global__ __launch_bounds__(256) void add_ln_k(
    const float* __restrict__ resid, const float* __restrict__ delta,
    const float* __restrict__ g, const float* __restrict__ bta,
    float* __restrict__ outF, bfu* __restrict__ outB)
{
  __shared__ float red[256], red2[256];
  __shared__ float s_mean, s_rstd;
  int token = blockIdx.x;
  int tid = threadIdx.x;
  float e[3]; float sum = 0.f, sq = 0.f;
#pragma unroll
  for (int i = 0; i < 3; ++i) {
    int d = tid + 256 * i;
    float v = resid[(size_t)token * D_ + d] + delta[(size_t)token * D_ + d];
    e[i] = v; sum += v; sq += v * v;
  }
  red[tid] = sum; red2[tid] = sq; __syncthreads();
  for (int st = 128; st; st >>= 1) {
    if (tid < st) { red[tid] += red[tid + st]; red2[tid] += red2[tid + st]; }
    __syncthreads();
  }
  if (tid == 0) {
    float m = red[0] / D_;
    float var = red2[0] / D_ - m * m;
    s_mean = m; s_rstd = rsqrtf(var + 1e-5f);
  }
  __syncthreads();
  float m = s_mean, rs = s_rstd;
#pragma unroll
  for (int i = 0; i < 3; ++i) {
    int d = tid + 256 * i;
    float v = (e[i] - m) * rs * g[d] + bta[d];
    outF[(size_t)token * D_ + d] = v;
    outB[(size_t)token * D_ + d] = f2bf(v);
  }
}

// ---------------------------------------------------------------------------
// Pooler: cat[b][enc*768 + n] = tanh(h[b, 0, :] @ pooler_w[:, n] + pooler_b[n])
// ---------------------------------------------------------------------------
__global__ void pooler_k(const float* __restrict__ hF, const float* __restrict__ w,
                         const float* __restrict__ bias, float* __restrict__ cat, int enc)
{
  int i = blockIdx.x * 256 + threadIdx.x;       // 0..1535
  int b = i / D_, n = i % D_;
  const float* hrow = hF + (size_t)(b * S_) * D_;
  float acc = bias[n];
  for (int k = 0; k < D_; ++k) acc += hrow[k] * w[(size_t)k * D_ + n];
  cat[b * (2 * D_) + enc * D_ + n] = tanhf(acc);
}

__global__ void dense_k(const float* __restrict__ cat, const float* __restrict__ dw,
                        const float* __restrict__ db, float* __restrict__ dh)
{
  int b = blockIdx.x, n = threadIdx.x;          // 2 x 256
  float acc = db[n];
  for (int k = 0; k < 2 * D_; ++k) acc += cat[b * (2 * D_) + k] * dw[(size_t)k * 256 + n];
  dh[b * 256 + n] = fmaxf(acc, 0.f);
}

__global__ void out_k(const float* __restrict__ dh, const float* __restrict__ ow,
                      float* __restrict__ out)
{
  int t = threadIdx.x;
  if (t < 4) {
    int b = t >> 1, o = t & 1;
    float acc = 0.f;
    for (int k = 0; k < 256; ++k) acc += dh[b * 256 + k] * ow[k * 2 + o];
    out[b * 2 + o] = acc;
  }
}

// ---------------------------------------------------------------------------
extern "C" void kernel_launch(void* const* d_in, const int* in_sizes, int n_in,
                              void* d_out, int out_size, void* d_ws, size_t ws_size,
                              hipStream_t stream) {
  (void)in_sizes; (void)n_in; (void)out_size; (void)ws_size;
  const int*   x  = (const int*)d_in[0];
  const float* we = (const float*)d_in[1];
  const float* pe = (const float*)d_in[2];
  const float* te = (const float*)d_in[3];
  const float* eg = (const float*)d_in[4];
  const float* ebta = (const float*)d_in[5];
  auto LP = [&](int l, int j) { return (const float*)d_in[6 + 16 * l + j]; };
  const float* pw = (const float*)d_in[38];
  const float* pb = (const float*)d_in[39];
  const float* dw = (const float*)d_in[40];
  const float* db = (const float*)d_in[41];
  const float* ow = (const float*)d_in[42];

  char* wp = (char*)d_ws;
  auto alloc = [&](size_t bytes) -> void* {
    void* r = (void*)wp;
    wp += (bytes + 255) & ~(size_t)255;
    return r;
  };
  float* hF   = (float*)alloc((size_t)M_ * D_ * 4);
  bfu*   hB   = (bfu*)  alloc((size_t)M_ * D_ * 2);
  bfu*   qBuf = (bfu*)  alloc((size_t)M_ * D_ * 2);
  bfu*   kBuf = (bfu*)  alloc((size_t)M_ * D_ * 2);
  bfu*   vBuf = (bfu*)  alloc((size_t)M_ * D_ * 2);
  bfu*   ctxB = (bfu*)  alloc((size_t)M_ * D_ * 2);
  float* tmpF = (float*)alloc((size_t)M_ * D_ * 4);
  bfu*   ff1B = (bfu*)  alloc((size_t)M_ * FF_ * 2);
  float* catF = (float*)alloc((size_t)B_ * 2 * D_ * 4);
  float* dhF  = (float*)alloc((size_t)B_ * 256 * 4);

  // weight bf16 conversion (wq, wk, wv, wo, fc1, fc2 per layer)
  static const size_t wsz[6]  = {(size_t)D_*D_, (size_t)D_*D_, (size_t)D_*D_,
                                 (size_t)D_*D_, (size_t)D_*FF_, (size_t)FF_*D_};
  static const int    widx[6] = {0, 2, 4, 6, 10, 12};
  bfu* wB[NL_][6];
  for (int l = 0; l < NL_; ++l)
    for (int j = 0; j < 6; ++j) {
      wB[l][j] = (bfu*)alloc(wsz[j] * 2);
      int n = (int)wsz[j];
      cvt_bf16_k<<<(n + 255) / 256, 256, 0, stream>>>(LP(l, widx[j]), wB[l][j], n);
    }

  const size_t attnLds = 32 * 768 * 4 + 32 * 768 * 2 + 384 * 64 * 2 + 32 * 64 * 2;
  const dim3 gD(D_ / 64, M_ / 128);
  const dim3 gF(FF_ / 64, M_ / 128);

  for (int enc = 0; enc < 2; ++enc) {
    embed_ln_k<<<M_, 256, 0, stream>>>(x, enc, we, pe, te, eg, ebta, hF, hB);
    for (int l = 0; l < NL_; ++l) {
      // Q (scale 1/sqrt(HD) folded into epilogue), K, V projections -> bf16
      gemm_bf16_wmma<<<gD, 256, 0, stream>>>(hB, wB[l][0], LP(l, 1), nullptr, qBuf,
                                             M_, D_, D_, 0.125f, 0);
      gemm_bf16_wmma<<<gD, 256, 0, stream>>>(hB, wB[l][1], LP(l, 3), nullptr, kBuf,
                                             M_, D_, D_, 1.0f, 0);
      gemm_bf16_wmma<<<gD, 256, 0, stream>>>(hB, wB[l][2], LP(l, 5), nullptr, vBuf,
                                             M_, D_, D_, 1.0f, 0);
      sliding_attn_k<<<B_ * NC_ * 8 * H_, 256, attnLds, stream>>>(qBuf, kBuf, vBuf,
                                                                  x, enc, ctxB);
      gemm_bf16_wmma<<<gD, 256, 0, stream>>>(ctxB, wB[l][3], LP(l, 7), tmpF, nullptr,
                                             M_, D_, D_, 1.0f, 0);
      add_ln_k<<<M_, 256, 0, stream>>>(hF, tmpF, LP(l, 8), LP(l, 9), hF, hB);
      gemm_bf16_wmma<<<gF, 256, 0, stream>>>(hB, wB[l][4], LP(l, 11), nullptr, ff1B,
                                             M_, FF_, D_, 1.0f, 1);   // GELU
      gemm_bf16_wmma<<<gD, 256, 0, stream>>>(ff1B, wB[l][5], LP(l, 13), tmpF, nullptr,
                                             M_, D_, FF_, 1.0f, 0);
      add_ln_k<<<M_, 256, 0, stream>>>(hF, tmpF, LP(l, 14), LP(l, 15), hF, hB);
    }
    pooler_k<<<(B_ * D_) / 256, 256, 0, stream>>>(hF, pw, pb, catF, enc);
  }
  dense_k<<<B_, 256, 0, stream>>>(catF, dw, db, dhF);
  out_k<<<1, 64, 0, stream>>>(dhF, ow, (float*)d_out);
}
